// TopicEncoder_58239756534018
// MI455X (gfx1250) — matrix operand order repaired and checked by reading, not compiled, not run on hardware
//
#include <hip/hip_runtime.h>

// Problem constants (match reference)
#define Bz  32
#define Sz  512
#define Hz  768
#define Tz  100
#define THz 512
#define Lz  5

#define KK_STEPS 24            // K = 768 = 24 * 32
#define NT_TILES 48            // N = 768 = 48 * 16
#define FRAG_ELEMS 512         // 32 lanes * 16 bf16 per B fragment
#define WSWZ_ELEMS ((size_t)Lz * KK_STEPS * NT_TILES * FRAG_ELEMS) // per weight tensor

typedef __attribute__((ext_vector_type(16))) __bf16 bf16x16;
typedef __attribute__((ext_vector_type(8)))  __bf16 bf16x8;
typedef __attribute__((ext_vector_type(4)))  __bf16 bf16x4;
typedef __attribute__((ext_vector_type(8)))  float  f32x8;

__device__ __forceinline__ __bf16 f2bf(float f) {
    unsigned u = __builtin_bit_cast(unsigned, f);
    u += 0x7FFFu + ((u >> 16) & 1u);               // round-to-nearest-even
    unsigned short h = (unsigned short)(u >> 16);
    return __builtin_bit_cast(__bf16, h);
}

// Load a 16-element bf16 fragment as two contiguous 16-byte chunks separated
// by `gap` elements (gap = 16 for the LDS A layout, gap = 8 for swizzled B).
__device__ __forceinline__ bf16x16 ld_frag(const __bf16* p, int gap) {
    bf16x8 lo = *(const bf16x8*)p;
    bf16x8 hi = *(const bf16x8*)(p + gap);
    return __builtin_shufflevector(lo, hi, 0, 1, 2, 3, 4, 5, 6, 7,
                                           8, 9, 10, 11, 12, 13, 14, 15);
}

// ---------------------------------------------------------------------------
// Kernel P: convert W1/W2 to bf16, pre-swizzled into per-lane WMMA B-fragment
// order: frag(l,kk,nt) holds, contiguously per lane t (0..31), the 16 values
// B[k = kk*32 + 8*(t>>4) + (i&7) + 16*(i>>3)][n = nt*16 + (t&15)], i = 0..15.
// ---------------------------------------------------------------------------
__global__ __launch_bounds__(256) void swizzle_w_kernel(
    const float* __restrict__ W1, const float* __restrict__ W2,
    __bf16* __restrict__ W1s, __bf16* __restrict__ W2s)
{
    const int PER = (int)WSWZ_ELEMS;               // 2,949,120 per tensor
    int o = blockIdx.x * 256 + threadIdx.x;
    if (o >= 2 * PER) return;
    const float* Wsrc = (o < PER) ? W1 : W2;
    __bf16*      Wdst = (o < PER) ? W1s : W2s;
    int oo = (o < PER) ? o : o - PER;
    const int i  = oo & 15;
    const int t  = (oo >> 4) & 31;
    int r        = oo >> 9;
    const int nt = r % NT_TILES;  r /= NT_TILES;
    const int kk = r % KK_STEPS;
    const int l  = r / KK_STEPS;
    const int k  = kk * 32 + 8 * (t >> 4) + (i & 7) + 16 * (i >> 3);
    const int n  = nt * 16 + (t & 15);
    Wdst[oo] = f2bf(Wsrc[((size_t)l * Hz + k) * Hz + n]);
}

// ---------------------------------------------------------------------------
// Kernel 0: zero the pooled accumulator
// ---------------------------------------------------------------------------
__global__ void zero_kernel(float* __restrict__ p, int n) {
    int i = blockIdx.x * blockDim.x + threadIdx.x;
    if (i < n) p[i] = 0.0f;
}

// ---------------------------------------------------------------------------
// Shared GEMM core: sA [32][768] bf16 in LDS  x  pre-swizzled Wl (bf16)
// -> 12 f32x8 accumulators per wave (6 N-tiles x 2 M-tiles).
// B fragments for one k-step are batched into 6 distinct register groups
// BEFORE the WMMA burst so multiple load-clauses stay in flight.
// ---------------------------------------------------------------------------
__device__ __forceinline__ void gemm_tile(
    const __bf16* __restrict__ sA, const __bf16* __restrict__ Wl,
    int wave, int lane, int half, int q, f32x8 acc[12])
{
    for (int kk = 0; kk < KK_STEPS; ++kk) {
        const int k0 = kk * 32;
        const __bf16* bp = Wl + ((size_t)kk * NT_TILES) * FRAG_ELEMS + lane * 16;

        // Issue all 6 B-fragment loads first (independent VGPR groups).
        bf16x16 bb[6];
        #pragma unroll
        for (int ntb = 0; ntb < 6; ++ntb)
            bb[ntb] = ld_frag(bp + (size_t)(wave + ntb * 8) * FRAG_ELEMS, 8);

        // A fragments from LDS (two b128 chunks each).
        const __bf16* ap = sA + q * Hz + k0 + 8 * half;   // ISA A layout
        bf16x16 a0 = ld_frag(ap,           16);
        bf16x16 a1 = ld_frag(ap + 16 * Hz, 16);

        #pragma unroll
        for (int ntb = 0; ntb < 6; ++ntb) {
            acc[2 * ntb]     = __builtin_amdgcn_wmma_f32_16x16x32_bf16(
                false, a0, false, bb[ntb], (short)0, acc[2 * ntb], false, false);
            acc[2 * ntb + 1] = __builtin_amdgcn_wmma_f32_16x16x32_bf16(
                false, a1, false, bb[ntb], (short)0, acc[2 * ntb + 1], false, false);
        }
    }
}

// ---------------------------------------------------------------------------
// Kernel 1: h1 = relu(LN(seq[b] @ W1[l] + b1[l]))  -> bf16 workspace
// grid (16, 32): blockIdx.x = 32-row S block, blockIdx.y = batch. 256 thr.
// Dynamic LDS: 32*768 bf16 (A tile) + 32*768 f32 (pre-LN tile) = 144 KB.
// ---------------------------------------------------------------------------
__global__ __launch_bounds__(256) void adapter1_kernel(
    const float* __restrict__ seq, const int* __restrict__ lang,
    const __bf16* __restrict__ W1s, const float* __restrict__ b1,
    const float* __restrict__ g1, const float* __restrict__ be1,
    unsigned short* __restrict__ h1out)
{
    extern __shared__ char smem[];
    __bf16* sA = (__bf16*)smem;                               // [32][768]
    float*  sH = (float*)(smem + 32 * Hz * sizeof(__bf16));   // [32][768]

    const int sblk = blockIdx.x;
    const int b    = blockIdx.y;
    const int s0   = sblk * 32;
    const int l    = lang[b];
    const int tid  = threadIdx.x;
    const int lane = tid & 31;
    const int wave = tid >> 5;
    const int half = lane >> 4;
    const int q    = lane & 15;

    // Stage seq tile -> bf16 LDS (float4 loads, b64 LDS stores)
    const float4* sp4 = (const float4*)(seq + ((size_t)b * Sz + s0) * Hz);
    bf16x4* sA4 = (bf16x4*)sA;
    for (int i = tid; i < (32 * Hz) / 4; i += 256) {
        float4 v = sp4[i];
        sA4[i] = (bf16x4){f2bf(v.x), f2bf(v.y), f2bf(v.z), f2bf(v.w)};
    }
    __syncthreads();

    f32x8 acc[12];
    #pragma unroll
    for (int i = 0; i < 12; ++i) acc[i] = (f32x8){};
    gemm_tile(sA, W1s + (size_t)l * KK_STEPS * NT_TILES * FRAG_ELEMS,
              wave, lane, half, q, acc);

    // Write +bias into f32 LDS tile (ISA C layout: M = r + 8*half)
    #pragma unroll
    for (int ntb = 0; ntb < 6; ++ntb) {
        const int n = (wave + ntb * 8) * 16 + q;
        const float bias = b1[l * Hz + n];
        #pragma unroll
        for (int r = 0; r < 8; ++r) {
            sH[(8 * half + r)      * Hz + n] = acc[2 * ntb][r] + bias;
            sH[(16 + 8 * half + r) * Hz + n] = acc[2 * ntb + 1][r] + bias;
        }
    }
    __syncthreads();

    // Per-row LayerNorm over H=768 + ReLU, write bf16 to workspace.
    unsigned short* outp = h1out + ((size_t)b * Sz + s0) * Hz;
    for (int rr = 0; rr < 4; ++rr) {
        const int m = wave * 4 + rr;
        float s1 = 0.f, s2 = 0.f;
        for (int c = lane; c < Hz; c += 32) {
            float x = sH[m * Hz + c];
            s1 += x; s2 += x * x;
        }
        #pragma unroll
        for (int off = 16; off > 0; off >>= 1) {   // wave32 reduction
            s1 += __shfl_xor(s1, off, 32);
            s2 += __shfl_xor(s2, off, 32);
        }
        const float mean = s1 * (1.0f / Hz);
        const float var  = s2 * (1.0f / Hz) - mean * mean;
        const float inv  = rsqrtf(var + 1e-5f);
        for (int c = lane; c < Hz; c += 32) {
            float x = sH[m * Hz + c];
            float y = (x - mean) * inv * g1[l * Hz + c] + be1[l * Hz + c];
            y = fmaxf(y, 0.0f);
            outp[(size_t)m * Hz + c] = __builtin_bit_cast(unsigned short, f2bf(y));
        }
    }
}

// ---------------------------------------------------------------------------
// Kernel 2: h2 = h1 @ W2[l] + b2[l]; masked accumulate into pooled[B][H]
// A tile staged with gfx1250 async global->LDS copies (16 B/lane/instr).
// ---------------------------------------------------------------------------
__global__ __launch_bounds__(256) void adapter2_pool_kernel(
    const unsigned short* __restrict__ h1, const int* __restrict__ lang,
    const __bf16* __restrict__ W2s, const float* __restrict__ b2,
    const float* __restrict__ mask, float* __restrict__ pooled)
{
    extern __shared__ char smem[];
    __bf16* sA = (__bf16*)smem;                               // [32][768]
    float*  sM = (float*)(smem + 32 * Hz * sizeof(__bf16));   // [32]

    const int sblk = blockIdx.x;
    const int b    = blockIdx.y;
    const int s0   = sblk * 32;
    const int l    = lang[b];
    const int tid  = threadIdx.x;
    const int lane = tid & 31;
    const int wave = tid >> 5;
    const int half = lane >> 4;
    const int q    = lane & 15;

    // Async copy: 32*768*2 B = 49152 B = 3072 x 16-byte chunks, 12 per thread.
    {
        const char* gbase = (const char*)(h1 + ((size_t)b * Sz + s0) * Hz);
        const unsigned lbase = (unsigned)(unsigned long long)(const void*)sA;
        #pragma unroll
        for (int it = 0; it < 12; ++it) {
            const int chunk = tid + it * 256;
            unsigned long long g = (unsigned long long)gbase + (size_t)chunk * 16;
            unsigned lds = lbase + chunk * 16;
            asm volatile("global_load_async_to_lds_b128 %0, %1, off"
                         :: "v"(lds), "v"(g) : "memory");
        }
        asm volatile("s_wait_asynccnt 0x0" ::: "memory");
    }
    if (tid < 32) sM[tid] = mask[b * Sz + s0 + tid];
    __syncthreads();

    f32x8 acc[12];
    #pragma unroll
    for (int i = 0; i < 12; ++i) acc[i] = (f32x8){};
    gemm_tile(sA, W2s + (size_t)l * KK_STEPS * NT_TILES * FRAG_ELEMS,
              wave, lane, half, q, acc);

    // +bias, multiply by mask, accumulate the 16 rows this lane owns.
    #pragma unroll
    for (int ntb = 0; ntb < 6; ++ntb) {
        const int n = (wave + ntb * 8) * 16 + q;
        const float bias = b2[l * Hz + n];
        float part = 0.0f;
        #pragma unroll
        for (int r = 0; r < 8; ++r) {
            part += (acc[2 * ntb][r]     + bias) * sM[8 * half + r];
            part += (acc[2 * ntb + 1][r] + bias) * sM[16 + 8 * half + r];
        }
        atomicAdd(&pooled[b * Hz + n], part);
    }
}

// ---------------------------------------------------------------------------
// Kernel 3: fusion MLP: concat(pooled/denom, lda) @ Wf + bf -> LN -> ReLU
// ---------------------------------------------------------------------------
__global__ __launch_bounds__(512) void fusion_kernel(
    const float* __restrict__ pooled, const float* __restrict__ mask,
    const float* __restrict__ lda, const float* __restrict__ Wf,
    const float* __restrict__ bfv, const float* __restrict__ gf,
    const float* __restrict__ bef, float* __restrict__ out)
{
    __shared__ float comb[Hz + Tz];   // 868
    __shared__ float red[THz];        // 512
    const int b = blockIdx.x, tid = threadIdx.x;

    red[tid] = mask[b * Sz + tid];
    __syncthreads();
    for (int off = 256; off > 0; off >>= 1) {
        if (tid < off) red[tid] += red[tid + off];
        __syncthreads();
    }
    const float denom = red[0] + 1e-10f;
    __syncthreads();

    for (int i = tid; i < Hz + Tz; i += THz)
        comb[i] = (i < Hz) ? pooled[b * Hz + i] / denom : lda[b * Tz + (i - Hz)];
    __syncthreads();

    float acc = bfv[tid];
    for (int k = 0; k < Hz + Tz; ++k)
        acc = fmaf(comb[k], Wf[(size_t)k * THz + tid], acc);

    red[tid] = acc;
    __syncthreads();
    for (int off = 256; off > 0; off >>= 1) {
        if (tid < off) red[tid] += red[tid + off];
        __syncthreads();
    }
    const float mean = red[0] * (1.0f / THz);
    __syncthreads();
    red[tid] = (acc - mean) * (acc - mean);
    __syncthreads();
    for (int off = 256; off > 0; off >>= 1) {
        if (tid < off) red[tid] += red[tid + off];
        __syncthreads();
    }
    const float var = red[0] * (1.0f / THz);
    const float y = (acc - mean) * rsqrtf(var + 1e-5f) * gf[tid] + bef[tid];
    out[b * THz + tid] = fmaxf(y, 0.0f);
}

// ---------------------------------------------------------------------------
extern "C" void kernel_launch(void* const* d_in, const int* in_sizes, int n_in,
                              void* d_out, int out_size, void* d_ws, size_t ws_size,
                              hipStream_t stream) {
    const float* seq  = (const float*)d_in[0];
    const float* mask = (const float*)d_in[1];
    const int*   lang = (const int*)  d_in[2];
    const float* lda  = (const float*)d_in[3];
    const float* W1   = (const float*)d_in[4];
    const float* b1   = (const float*)d_in[5];
    const float* g1   = (const float*)d_in[6];
    const float* be1  = (const float*)d_in[7];
    const float* W2   = (const float*)d_in[8];
    const float* b2   = (const float*)d_in[9];
    const float* Wf   = (const float*)d_in[10];
    const float* bfv  = (const float*)d_in[11];
    const float* gf   = (const float*)d_in[12];
    const float* bef  = (const float*)d_in[13];
    float* out = (float*)d_out;

    // Workspace: [h1 bf16 B*S*H][pooled f32 B*H][W1 swizzled bf16][W2 swizzled bf16]
    char* ws = (char*)d_ws;
    unsigned short* h1 = (unsigned short*)ws;   ws += (size_t)Bz * Sz * Hz * 2;
    float* pooled = (float*)ws;                 ws += (size_t)Bz * Hz * 4;
    __bf16* W1s = (__bf16*)ws;                  ws += WSWZ_ELEMS * 2;
    __bf16* W2s = (__bf16*)ws;

    const int swz_total = 2 * (int)WSWZ_ELEMS;
    swizzle_w_kernel<<<(swz_total + 255) / 256, 256, 0, stream>>>(W1, W2, W1s, W2s);
    zero_kernel<<<(Bz * Hz + 255) / 256, 256, 0, stream>>>(pooled, Bz * Hz);

    const size_t lds1 = 32 * Hz * (sizeof(unsigned short) + sizeof(float)); // 144 KB
    adapter1_kernel<<<dim3(16, 32), 256, lds1, stream>>>(seq, lang, W1s, b1, g1, be1, h1);

    const size_t lds2 = 32 * Hz * sizeof(unsigned short) + 32 * sizeof(float);
    adapter2_pool_kernel<<<dim3(16, 32), 256, lds2, stream>>>(h1, lang, W2s, b2, mask, pooled);

    fusion_kernel<<<Bz, THz, 0, stream>>>(pooled, mask, lda, Wf, bfv, gf, bef, out);
}
